// AttentionSequencePoolingLayer_26207890440828
// MI455X (gfx1250) — matrix operand order repaired and checked by reading, not compiled, not run on hardware
//
#include <hip/hip_runtime.h>
#include <hip/hip_bf16.h>

typedef __attribute__((ext_vector_type(16))) __bf16        v16bf;
typedef __attribute__((ext_vector_type(8)))  float         v8f;
typedef __attribute__((ext_vector_type(4)))  float         v4f;
typedef __attribute__((ext_vector_type(4)))  unsigned int  v4u;

#define EDIM 64
#define TLEN 200
#define TPAD 208      // 13*16
#define H1   80
#define H2   40
#define H2P  48       // padded N for gemm2
#define H1S  80       // h1 row stride in elements (160B)

union Frag {
    v4u    u[2];
    v16bf  v;
    __bf16 h[16];
};

__device__ __forceinline__ __bf16 tobf(float f) { return (__bf16)f; }

__device__ __forceinline__ float sigmoidf(float x) {
    // fast: v_exp_f32 + v_rcp_f32 (reference precision is bf16-limited anyway)
    return __builtin_amdgcn_rcpf(1.0f + __expf(-x));
}

__global__ __launch_bounds__(256)
void din_attn_kernel(const float* __restrict__ query,
                     const float* __restrict__ keys,
                     const int*   __restrict__ keys_length,
                     const float* __restrict__ W1, const float* __restrict__ b1,
                     const float* __restrict__ W2, const float* __restrict__ b2,
                     const float* __restrict__ W3, const float* __restrict__ b3,
                     float* __restrict__ out)
{
    __shared__ __align__(16) __bf16 sWeffT[H1 * EDIM];   // [80][64] bf16, N-major
    __shared__ __align__(16) __bf16 sH1[TPAD * H1S];     // [208][80] bf16
    __shared__ __align__(16) __bf16 sW2T[H2P * H1S];     // [48][80] bf16, N-major
    __shared__ float sQ[EDIM];
    __shared__ float sC[H1];
    __shared__ float sB2[H2];
    __shared__ float sW3[H2];
    __shared__ float sScore[TPAD];
    __shared__ float sWt[TPAD];
    __shared__ float sRed[256];

    const int tid  = threadIdx.x;
    const int lane = tid & 31;
    const int wave = tid >> 5;
    const int half = (lane >> 4) & 1;   // which 16-lane half
    const int lm   = lane & 15;
    const int b    = blockIdx.x;
    const float* kb = keys + (long)b * (TLEN * EDIM);

    // ---- phase 0: per-batch setup -------------------------------------
    if (tid < EDIM) sQ[tid] = query[b * EDIM + tid];
    __syncthreads();

    // WeffT[j][e] = W1[64+e][j] - W1[128+e][j] + q[e]*W1[192+e][j]
    for (int idx = tid; idx < H1 * EDIM; idx += 256) {
        int j = idx % H1, e = idx / H1;
        float v = W1[(EDIM + e) * H1 + j] - W1[(2 * EDIM + e) * H1 + j]
                + sQ[e] * W1[(3 * EDIM + e) * H1 + j];
        sWeffT[j * EDIM + e] = tobf(v);
    }
    // c[j] = b1[j] + sum_e q[e]*(W1[e][j] + W1[128+e][j])
    if (tid < H1) {
        float c = b1[tid];
        for (int e = 0; e < EDIM; ++e)
            c += sQ[e] * (W1[e * H1 + tid] + W1[(2 * EDIM + e) * H1 + tid]);
        sC[tid] = c;
    }
    // W2T[j][k] = W2[k][j], rows 40..47 zero
    for (int idx = tid; idx < H2P * H1; idx += 256) {
        int j = idx % H2P, k = idx / H2P;
        float v = (j < H2) ? W2[k * H2 + j] : 0.0f;
        sW2T[j * H1S + k] = tobf(v);
    }
    if (tid < H2) { sB2[tid] = b2[tid]; sW3[tid] = W3[tid]; }
    if (tid < TPAD) sScore[tid] = 0.0f;
    __syncthreads();

    // ---- GEMM1: h1 = sigmoid(keys @ Weff + c) -------------------------
    // per-wave m-tiles; A converted once, reused across all 5 n-tiles
    for (int tm = wave; tm < 13; tm += 8) {
        int row  = tm * 16 + lm;
        int rowc = row < TLEN ? row : (TLEN - 1);   // clamp; padded rows unused
        const float* ap = kb + rowc * EDIM;
        Frag aF[2];
        #pragma unroll
        for (int kc = 0; kc < 2; ++kc) {
            // lane holds K = {0..7, 16..23} (+8 for upper half) of 32-chunk
            const float* p = ap + kc * 32 + half * 8;
            v4f f0 = *(const v4f*)(p + 0);
            v4f f1 = *(const v4f*)(p + 4);
            v4f f2 = *(const v4f*)(p + 16);
            v4f f3 = *(const v4f*)(p + 20);
            aF[kc].h[0]=tobf(f0.x);  aF[kc].h[1]=tobf(f0.y);
            aF[kc].h[2]=tobf(f0.z);  aF[kc].h[3]=tobf(f0.w);
            aF[kc].h[4]=tobf(f1.x);  aF[kc].h[5]=tobf(f1.y);
            aF[kc].h[6]=tobf(f1.z);  aF[kc].h[7]=tobf(f1.w);
            aF[kc].h[8]=tobf(f2.x);  aF[kc].h[9]=tobf(f2.y);
            aF[kc].h[10]=tobf(f2.z); aF[kc].h[11]=tobf(f2.w);
            aF[kc].h[12]=tobf(f3.x); aF[kc].h[13]=tobf(f3.y);
            aF[kc].h[14]=tobf(f3.z); aF[kc].h[15]=tobf(f3.w);
        }
        for (int tn = 0; tn < 5; ++tn) {
            v8f acc = {};
            #pragma unroll
            for (int kc = 0; kc < 2; ++kc) {
                // B: lane = column n, K = {0..7, 16..23} (+8 for upper half)
                Frag bf;
                const v4u* bp = (const v4u*)&sWeffT[(tn * 16 + lm) * EDIM
                                                    + kc * 32 + half * 8];
                bf.u[0] = bp[0];
                bf.u[1] = bp[2];
                acc = __builtin_amdgcn_wmma_f32_16x16x32_bf16(
                          false, aF[kc].v, false, bf.v, (short)0, acc, false, false);
            }
            int n = tn * 16 + lm;
            float cb = sC[n];
            #pragma unroll
            for (int r = 0; r < 8; ++r) {
                int m = tm * 16 + r + half * 8;
                sH1[m * H1S + n] = tobf(sigmoidf(acc[r] + cb));
            }
        }
    }
    __syncthreads();

    // ---- GEMM2 + fold W3: score[m] += sigmoid(h1@W2 + b2)[n] * W3[n] --
    for (int tm = wave; tm < 13; tm += 8) {
        int arow = tm * 16 + lm;
        v4u z = {0u, 0u, 0u, 0u};
        Frag aF[3];
        #pragma unroll
        for (int kc = 0; kc < 2; ++kc) {
            const v4u* ap2 = (const v4u*)&sH1[arow * H1S + kc * 32 + half * 8];
            aF[kc].u[0] = ap2[0];
            aF[kc].u[1] = ap2[2];                // +32 bytes
        }
        {   // K 64..79 real, 80..95 zero
            const v4u* ap2 = (const v4u*)&sH1[arow * H1S + 64 + half * 8];
            aF[2].u[0] = ap2[0];
            aF[2].u[1] = z;
        }
        for (int tn = 0; tn < 3; ++tn) {
            int j = tn * 16 + lm;
            v8f acc = {};
            #pragma unroll
            for (int kc = 0; kc < 3; ++kc) {
                Frag bf;
                if (kc < 2) {
                    const v4u* bp = (const v4u*)&sW2T[j * H1S + kc * 32 + half * 8];
                    bf.u[0] = bp[0];
                    bf.u[1] = bp[2];
                } else {
                    const v4u* bp = (const v4u*)&sW2T[j * H1S + 64 + half * 8];
                    v4u b0 = bp[0], b1v = z;   // K 80..95 -> 0
                    bf.u[0] = b0;
                    bf.u[1] = b1v;
                }
                acc = __builtin_amdgcn_wmma_f32_16x16x32_bf16(
                          false, aF[kc].v, false, bf.v, (short)0, acc, false, false);
            }
            int n = tn * 16 + lm;
            if (n < H2) {
                float w3n = sW3[n], bias = sB2[n];
                #pragma unroll
                for (int r = 0; r < 8; ++r) {
                    int m = tm * 16 + r + half * 8;
                    atomicAdd(&sScore[m], sigmoidf(acc[r] + bias) * w3n);  // ds_add_f32
                }
            }
        }
    }
    __syncthreads();

    // ---- masked softmax over T=200 ------------------------------------
    const float NEGV = -4294967295.0f;          // -2^32 + 1 (ref padding)
    int len = keys_length[b];
    float sval = -INFINITY;
    if (tid < TLEN) sval = (tid < len) ? (sScore[tid] + b3[0]) : NEGV;
    sRed[tid] = sval;
    __syncthreads();
    for (int s = 128; s > 0; s >>= 1) {
        if (tid < s) sRed[tid] = fmaxf(sRed[tid], sRed[tid + s]);
        __syncthreads();
    }
    float mx = sRed[0];
    __syncthreads();
    float ev = (tid < TLEN) ? __expf(sval - mx) : 0.0f;
    sRed[tid] = ev;
    __syncthreads();
    for (int s = 128; s > 0; s >>= 1) {
        if (tid < s) sRed[tid] += sRed[tid + s];
        __syncthreads();
    }
    float inv = __builtin_amdgcn_rcpf(sRed[0]);
    __syncthreads();
    if (tid < TPAD) sWt[tid] = (tid < TLEN) ? ev * inv : 0.0f;
    __syncthreads();

    // ---- out[b][e] = sum_t w[t] * keys[b][t][e] -----------------------
    int ee = tid & 63, g = tid >> 6;
    float accv = 0.0f;
    for (int t = g; t < TLEN; t += 4)
        accv += sWt[t] * kb[t * EDIM + ee];
    sRed[tid] = accv;
    __syncthreads();
    if (tid < EDIM)
        out[b * EDIM + tid] = sRed[tid] + sRed[tid + 64] + sRed[tid + 128] + sRed[tid + 192];
}

extern "C" void kernel_launch(void* const* d_in, const int* in_sizes, int n_in,
                              void* d_out, int out_size, void* d_ws, size_t ws_size,
                              hipStream_t stream) {
    const float* query = (const float*)d_in[0];
    const float* keys  = (const float*)d_in[1];
    const int*   klen  = (const int*)  d_in[2];
    const float* W1    = (const float*)d_in[3];
    const float* b1    = (const float*)d_in[4];
    const float* W2    = (const float*)d_in[5];
    const float* b2    = (const float*)d_in[6];
    const float* W3    = (const float*)d_in[7];
    const float* b3    = (const float*)d_in[8];
    din_attn_kernel<<<4096, 256, 0, stream>>>(query, keys, klen,
                                              W1, b1, W2, b2, W3, b3,
                                              (float*)d_out);
}
